// MultiHeadAttention_74423193305634
// MI455X (gfx1250) — matrix operand order repaired and checked by reading, not compiled, not run on hardware
//
#include <hip/hip_runtime.h>
#include <hip/hip_bf16.h>
#include <math.h>

// Problem constants (reference: B=2, S=2048, HID=2048, H=16, D=128)
#define Bsz  2
#define Sq   2048
#define HIDD 2048
#define NH   16
#define HD   128
#define Mrows (Bsz * Sq)          // 4096 token rows

typedef __bf16 bf16;
typedef __bf16 v16bf __attribute__((ext_vector_type(16)));
typedef float  v8f   __attribute__((ext_vector_type(8)));
typedef unsigned int u32x4 __attribute__((ext_vector_type(4)));

union FragU { v16bf v; u32x4 q[2]; };

// Load a 16-element bf16 fragment as two 16B (b128) chunks.
__device__ __forceinline__ v16bf load_frag(const bf16* p0, const bf16* p1) {
  FragU f;
  f.q[0] = *reinterpret_cast<const u32x4*>(p0);
  f.q[1] = *reinterpret_cast<const u32x4*>(p1);
  return f.v;
}

__device__ __forceinline__ v8f wmma_bf16(v16bf a, v16bf b, v8f c) {
  // (neg_a, A, neg_b, B, c_mod, C, reuse_a, reuse_b)
  return __builtin_amdgcn_wmma_f32_16x16x32_bf16(false, a, false, b, (short)0, c,
                                                 false, false);
}

// ---------------------------------------------------------------------------
// fp32 -> bf16 conversion (vectorized x4)
// ---------------------------------------------------------------------------
__global__ void cvt_f32_to_bf16(const float* __restrict__ in,
                                bf16* __restrict__ out, int n) {
  int i = (blockIdx.x * blockDim.x + threadIdx.x) * 4;
  if (i < n) {
    float4 v = *reinterpret_cast<const float4*>(in + i);
    out[i + 0] = (bf16)v.x;
    out[i + 1] = (bf16)v.y;
    out[i + 2] = (bf16)v.z;
    out[i + 3] = (bf16)v.w;
  }
}

// ---------------------------------------------------------------------------
// GEMM: C[M,N] = A[M,K] * W[N,K]^T   (A,W bf16; acc f32; out f32 or bf16)
// Block = 128 thr (4 waves), block tile 64x128.
// Waves in 2x2 grid; each wave computes 32x64 (2 M-tiles x 4 N-tiles):
//   8 WMMAs per k-step vs 12 b128 loads (1.5 loads/WMMA).
// ---------------------------------------------------------------------------
__global__ __launch_bounds__(128)
void gemm_bf16(const bf16* __restrict__ A, const bf16* __restrict__ W,
               void* __restrict__ Cout, int M, int N, int Kd, int store_bf16) {
  const int lane = threadIdx.x & 31;
  const int w    = threadIdx.x >> 5;
  const int g    = lane >> 4;      // K-half select
  const int l16  = lane & 15;

  const int bm = blockIdx.y * 64  + (w >> 1) * 32;
  const int bn = blockIdx.x * 128 + (w & 1) * 64;

  v8f acc[2][4];
#pragma unroll
  for (int mi = 0; mi < 2; ++mi)
#pragma unroll
    for (int ni = 0; ni < 4; ++ni) acc[mi][ni] = (v8f)0.0f;

  const bf16* Ar[2];
#pragma unroll
  for (int mi = 0; mi < 2; ++mi)
    Ar[mi] = A + (size_t)(bm + mi * 16 + l16) * Kd;      // A-frag rows (m = l16)
  const bf16* Wr[4];
#pragma unroll
  for (int ni = 0; ni < 4; ++ni)
    Wr[ni] = W + (size_t)(bn + ni * 16 + l16) * Kd;      // B-frag cols (n = l16)

  const int ksA = g * 8;    // A: elems0-7 -> K=ksA.. ; elems8-15 -> K=16+ksA..
  const int ksB = g * 16;   // B-frag loads use ksA (same k order as A); ksB only offsets the prefetch

  for (int kk = 0; kk < Kd; kk += 32) {
    // Pull the tiles 8 k-steps ahead toward the WGP (global_prefetch_b8).
    if (kk + 256 < Kd) {
      __builtin_prefetch(Ar[0] + kk + 256 + ksA, 0, 1);
      __builtin_prefetch(Wr[0] + kk + 256 + ksB, 0, 1);
      __builtin_prefetch(Wr[2] + kk + 256 + ksB, 0, 1);
    }
    v16bf af[2], bftile[4];
#pragma unroll
    for (int mi = 0; mi < 2; ++mi)
      af[mi] = load_frag(Ar[mi] + kk + ksA, Ar[mi] + kk + 16 + ksA);
#pragma unroll
    for (int ni = 0; ni < 4; ++ni)
      bftile[ni] = load_frag(Wr[ni] + kk + ksA, Wr[ni] + kk + 16 + ksA);
#pragma unroll
    for (int mi = 0; mi < 2; ++mi)
#pragma unroll
      for (int ni = 0; ni < 4; ++ni)
        acc[mi][ni] = wmma_bf16(af[mi], bftile[ni], acc[mi][ni]);
  }

#pragma unroll
  for (int mi = 0; mi < 2; ++mi)
#pragma unroll
    for (int ni = 0; ni < 4; ++ni)
#pragma unroll
      for (int v = 0; v < 8; ++v) {
        int row = bm + mi * 16 + v + 8 * g;   // C/D layout: M = v + 8*(lane>=16)
        int col = bn + ni * 16 + l16;
        float val = acc[mi][ni][v];
        if (store_bf16)
          ((bf16*)Cout)[(size_t)row * N + col] = (bf16)val;
        else
          ((float*)Cout)[(size_t)row * N + col] = val;
      }
}

// ---------------------------------------------------------------------------
// RoPE + relayout. mode 0: Q (scale 1/sqrt(D)) -> [B,H,S,D]
//                  mode 1: K                  -> [B,H,S,D]
//                  mode 2: V (no rope)        -> [B,H,D,S] (transposed)
// Grid (S, H, B), block 128 (= D threads).
// ---------------------------------------------------------------------------
__global__ __launch_bounds__(128)
void rope_transpose(const bf16* __restrict__ src, const float* __restrict__ cs,
                    const float* __restrict__ sn, bf16* __restrict__ dst,
                    int mode) {
  const int s = blockIdx.x, h = blockIdx.y, b = blockIdx.z;
  const int d = threadIdx.x;
  const size_t si = ((size_t)(b * Sq + s) * NH + h) * HD + d;
  float x = (float)src[si];
  if (mode == 2) {
    dst[((size_t)(b * NH + h) * HD + d) * Sq + s] = (bf16)x;
    return;
  }
  const size_t ci = (size_t)(b * Sq + s) * HD + d;
  float c = cs[ci], si_ = sn[ci];
  float part = (d < 64) ? -(float)src[si + 64] : (float)src[si - 64];
  float r = x * c + part * si_;
  if (mode == 0) r *= 0.08838834764831845f;   // 1/sqrt(128)
  dst[((size_t)(b * NH + h) * Sq + s) * HD + d] = (bf16)r;
}

// ---------------------------------------------------------------------------
// Flash attention (causal, online softmax). Grid (S/64, H, B), block 128.
// Wave w owns query rows qr0..qr0+15; streams key blocks of 32.
// Q,K: [B,H,S,D] bf16 (Q pre-scaled).  V: [B,H,D,S] bf16.  Ctx: [B,S,H,D] bf16.
// ---------------------------------------------------------------------------
__global__ __launch_bounds__(128)
void flash_attn(const bf16* __restrict__ Qh, const bf16* __restrict__ Kh,
                const bf16* __restrict__ Vt, bf16* __restrict__ Ctx) {
  __shared__ float ldsP[4][16][32];   // per-wave P tile (16 rows x 32 keys)

  const int qb = blockIdx.x, h = blockIdx.y, b = blockIdx.z;
  const int w = threadIdx.x >> 5, lane = threadIdx.x & 31;
  const int g = lane >> 4, l16 = lane & 15;
  const int qr0 = qb * 64 + w * 16;

  const bf16* qp = Qh + (size_t)(b * NH + h) * Sq * HD;
  const bf16* kp = Kh + (size_t)(b * NH + h) * Sq * HD;
  const bf16* vp = Vt + (size_t)(b * NH + h) * HD * Sq;

  // Q fragments: 16 rows x 128 cols = 4 A-fragments (K chunks of 32)
  v16bf qa[4];
  const bf16* qrow = qp + (size_t)(qr0 + l16) * HD;
#pragma unroll
  for (int c = 0; c < 4; ++c)
    qa[c] = load_frag(qrow + c * 32 + g * 8, qrow + c * 32 + 16 + g * 8);

  v8f o[8];
#pragma unroll
  for (int t = 0; t < 8; ++t) o[t] = (v8f)0.0f;
  float mrow[8], lrow[8];
#pragma unroll
  for (int v = 0; v < 8; ++v) { mrow[v] = -INFINITY; lrow[v] = 0.0f; }

  for (int kb = 0; kb <= qr0 + 15; kb += 32) {
    // ---- scores: two 16x16 tiles over keys kb..kb+15 and kb+16..kb+31 ----
    v8f s0 = (v8f)0.0f, s1 = (v8f)0.0f;
    const bf16* kr0 = kp + (size_t)(kb + l16) * HD;
    const bf16* kr1 = kp + (size_t)(kb + 16 + l16) * HD;
#pragma unroll
    for (int c = 0; c < 4; ++c) {
      v16bf k0 = load_frag(kr0 + c * 32 + g * 8, kr0 + c * 32 + 16 + g * 8);
      v16bf k1 = load_frag(kr1 + c * 32 + g * 8, kr1 + c * 32 + 16 + g * 8);
      s0 = wmma_bf16(qa[c], k0, s0);
      s1 = wmma_bf16(qa[c], k1, s1);
    }

    // ---- causal mask on diagonal blocks ----
    if (kb + 31 > qr0) {
#pragma unroll
      for (int v = 0; v < 8; ++v) {
        int row = qr0 + v + 8 * g;
        if (kb + l16 > row)      s0[v] = -INFINITY;
        if (kb + 16 + l16 > row) s1[v] = -INFINITY;
      }
    }

    // ---- online softmax (row stats replicated across each 16-lane group) --
#pragma unroll
    for (int v = 0; v < 8; ++v) {
      float mx = fmaxf(s0[v], s1[v]);
#pragma unroll
      for (int m = 8; m >= 1; m >>= 1) mx = fmaxf(mx, __shfl_xor(mx, m, 32));
      float newm = fmaxf(mrow[v], mx);
      float corr = __expf(mrow[v] - newm);
      float p0 = __expf(s0[v] - newm);
      float p1 = __expf(s1[v] - newm);
      float ps = p0 + p1;
#pragma unroll
      for (int m = 8; m >= 1; m >>= 1) ps += __shfl_xor(ps, m, 32);
      lrow[v] = lrow[v] * corr + ps;
      mrow[v] = newm;
#pragma unroll
      for (int t = 0; t < 8; ++t) o[t][v] *= corr;
      // stage P in C-layout into per-wave LDS tile
      ldsP[w][v + 8 * g][l16]      = p0;
      ldsP[w][v + 8 * g][16 + l16] = p1;
    }
    asm volatile("s_wait_dscnt 0" ::: "memory");

    // ---- read P back in A-fragment layout (keys are the contraction dim) --
    v16bf pv;
    const float* pr = &ldsP[w][l16][0];
#pragma unroll
    for (int i = 0; i < 8; ++i) {
      pv[i]     = (bf16)pr[g * 8 + i];
      pv[8 + i] = (bf16)pr[16 + g * 8 + i];
    }

    // ---- O += P * V  (V transposed: contiguous along keys) ----
#pragma unroll
    for (int t = 0; t < 8; ++t) {
      const bf16* vc = vp + (size_t)(t * 16 + l16) * Sq + kb + g * 8;
      v16bf vf = load_frag(vc, vc + 16);
      o[t] = wmma_bf16(pv, vf, o[t]);
    }
  }

  // ---- normalize and store context [B,S,H,D] ----
#pragma unroll
  for (int v = 0; v < 8; ++v) {
    float inv = 1.0f / lrow[v];
    int srow = qr0 + v + 8 * g;
    size_t base = ((size_t)(b * Sq + srow) * NH + h) * HD;
#pragma unroll
    for (int t = 0; t < 8; ++t)
      Ctx[base + t * 16 + l16] = (bf16)(o[t][v] * inv);
  }
}

// ---------------------------------------------------------------------------
extern "C" void kernel_launch(void* const* d_in, const int* in_sizes, int n_in,
                              void* d_out, int out_size, void* d_ws, size_t ws_size,
                              hipStream_t stream) {
  const float* hs   = (const float*)d_in[0];  // [B,S,HID]
  const float* cosb = (const float*)d_in[1];  // [B,S,D]
  const float* sinb = (const float*)d_in[2];  // [B,S,D]
  const float* wq   = (const float*)d_in[3];  // [H*D,HID]
  const float* wk   = (const float*)d_in[4];
  const float* wv   = (const float*)d_in[5];
  const float* wo   = (const float*)d_in[6];  // [HID,H*D]
  // d_in[7] = position_ids (identity; unused)

  const size_t XB  = (size_t)Mrows * HIDD * sizeof(bf16);   // 16.8 MB
  const size_t WB  = (size_t)HIDD * HIDD * sizeof(bf16);    //  8.4 MB

  char* ws = (char*)d_ws;
  size_t off = 0;
  auto take = [&](size_t bytes) { char* p = ws + off; off += (bytes + 255) & ~(size_t)255; return p; };

  bf16* Xbf = (bf16*)take(XB);
  bf16* Wqb = (bf16*)take(WB);
  bf16* Wkb = (bf16*)take(WB);
  bf16* Wvb = (bf16*)take(WB);
  bf16* Wob = (bf16*)take(WB);
  bf16* Qp  = (bf16*)take(XB);
  bf16* Kp  = (bf16*)take(XB);
  bf16* Vp  = (bf16*)take(XB);
  bf16* Kh  = (bf16*)take(XB);
  // Aliases (lifetime-disjoint): X dead after QKV gemms; Qp dead after rope-Q;
  // Kp dead after rope-K.
  bf16* Qh  = Xbf;
  bf16* Vt  = Qp;
  bf16* Ctx = Kp;

  // 1) fp32 -> bf16
  {
    int nX = Mrows * HIDD;          // 8,388,608
    int nW = HIDD * HIDD;           // 4,194,304
    cvt_f32_to_bf16<<<nX / 4 / 256, 256, 0, stream>>>(hs, Xbf, nX);
    cvt_f32_to_bf16<<<nW / 4 / 256, 256, 0, stream>>>(wq, Wqb, nW);
    cvt_f32_to_bf16<<<nW / 4 / 256, 256, 0, stream>>>(wk, Wkb, nW);
    cvt_f32_to_bf16<<<nW / 4 / 256, 256, 0, stream>>>(wv, Wvb, nW);
    cvt_f32_to_bf16<<<nW / 4 / 256, 256, 0, stream>>>(wo, Wob, nW);
  }

  // 2) QKV projections (bf16 out)
  dim3 gg(HIDD / 128, Mrows / 64);
  gemm_bf16<<<gg, 128, 0, stream>>>(Xbf, Wqb, Qp, Mrows, HIDD, HIDD, 1);
  gemm_bf16<<<gg, 128, 0, stream>>>(Xbf, Wkb, Kp, Mrows, HIDD, HIDD, 1);
  gemm_bf16<<<gg, 128, 0, stream>>>(Xbf, Wvb, Vp, Mrows, HIDD, HIDD, 1);

  // 3) RoPE + relayout
  dim3 gr(Sq, NH, Bsz);
  rope_transpose<<<gr, 128, 0, stream>>>(Qp, cosb, sinb, Qh, 0);
  rope_transpose<<<gr, 128, 0, stream>>>(Kp, cosb, sinb, Kh, 1);
  rope_transpose<<<gr, 128, 0, stream>>>(Vp, cosb, sinb, Vt, 2);

  // 4) causal flash attention
  dim3 ga(Sq / 64, NH, Bsz);
  flash_attn<<<ga, 128, 0, stream>>>(Qh, Kh, Vt, Ctx);

  // 5) output projection (f32 out)
  gemm_bf16<<<gg, 128, 0, stream>>>(Ctx, Wob, (float*)d_out, Mrows, HIDD, HIDD, 0);
}
